// Block_463856467972
// MI455X (gfx1250) — matrix operand order repaired and checked by reading, not compiled, not run on hardware
//
#include <hip/hip_runtime.h>
#include <hip/hip_bf16.h>
#include <cstdint>

// ---------- constants for this problem ----------
static constexpr int BB   = 4;
static constexpr int TT   = 2048;
static constexpr int CC   = 1024;
static constexpr int HH   = 16;
static constexpr int HS   = 64;
static constexpr int C3   = 3 * CC;     // 3072
static constexpr int FF   = 4 * CC;     // 4096
static constexpr int MM   = BB * TT;    // 8192 rows

typedef __bf16 bf16;
typedef __attribute__((ext_vector_type(16))) __bf16 v16bf;
typedef __attribute__((ext_vector_type(8)))  __bf16 v8bf;
typedef __attribute__((ext_vector_type(8)))  float  v8f;

union V16 { v16bf v; v8bf h[2]; };

__device__ __forceinline__ v8f wmma_bf16(v16bf a, v16bf b, v8f c) {
  // D = A(16x32 bf16) * B(32x16 bf16) + C(16x16 f32)
  return __builtin_amdgcn_wmma_f32_16x16x32_bf16(false, a, false, b, (short)0, c,
                                                 false, false);
}

// CDNA5 async global->LDS copy (ASYNCcnt-tracked). vdst carries the LDS byte
// address; vaddr is the 64-bit global address.  (CDNA5 ISA)
__device__ __forceinline__ void async_b128(const bf16* g, uint32_t lds_byte_off) {
  asm volatile("global_load_async_to_lds_b128 %0, %1, off"
               :: "v"(lds_byte_off), "v"((unsigned long long)(uintptr_t)g)
               : "memory");
}
__device__ __forceinline__ uint32_t lds_off(const void* p) {
  // generic LDS pointer keeps the LDS byte offset in addr[31:0]
  return (uint32_t)(uintptr_t)p;
}

// ---------------- LayerNorm -> bf16 ----------------
__global__ __launch_bounds__(256)
void layernorm_bf16_kernel(const float* __restrict__ x, const float* __restrict__ g,
                           const float* __restrict__ bb, bf16* __restrict__ out) {
  const int C = CC;
  int row = blockIdx.x;
  const float* xr = x + (size_t)row * C;
  __shared__ float red[8];
  __shared__ float s_mu, s_rs;
  float lv[4];
  float s = 0.f;
#pragma unroll
  for (int i = 0; i < 4; ++i) { lv[i] = xr[threadIdx.x + i * 256]; s += lv[i]; }
  for (int off = 16; off; off >>= 1) s += __shfl_xor(s, off, 32);
  if ((threadIdx.x & 31) == 0) red[threadIdx.x >> 5] = s;
  __syncthreads();
  if (threadIdx.x == 0) {
    float t = 0.f;
#pragma unroll
    for (int w = 0; w < 8; ++w) t += red[w];
    s_mu = t * (1.0f / C);
  }
  __syncthreads();
  float mu = s_mu;
  float v = 0.f;
#pragma unroll
  for (int i = 0; i < 4; ++i) { float d = lv[i] - mu; v += d * d; }
  for (int off = 16; off; off >>= 1) v += __shfl_xor(v, off, 32);
  if ((threadIdx.x & 31) == 0) red[threadIdx.x >> 5] = v;
  __syncthreads();
  if (threadIdx.x == 0) {
    float t = 0.f;
#pragma unroll
    for (int w = 0; w < 8; ++w) t += red[w];
    s_rs = rsqrtf(t * (1.0f / C) + 1e-5f);
  }
  __syncthreads();
  float rs = s_rs;
#pragma unroll
  for (int i = 0; i < 4; ++i) {
    int c = threadIdx.x + i * 256;
    out[(size_t)row * C + c] = (bf16)((lv[i] - mu) * rs * g[c] + bb[c]);
  }
}

// ---------------- weight prep ----------------
// Pack Wq/Wk/Wv [H,C,hs] into one transposed bf16 matrix [3C, C]:
// out[(s*C + h*64 + d)*C + c] = W_s[h][c][d]
__global__ __launch_bounds__(256)
void pack_qkv_w_kernel(const float* __restrict__ Wq, const float* __restrict__ Wk,
                       const float* __restrict__ Wv, bf16* __restrict__ out) {
  size_t idx = (size_t)blockIdx.x * 256 + threadIdx.x;
  if (idx >= (size_t)C3 * CC) return;
  int c = (int)(idx & (CC - 1));
  int rowOut = (int)(idx >> 10);       // 0..3071
  int s = rowOut >> 10;                // which of q/k/v
  int hd = rowOut & (CC - 1);          // h*64+d
  int h = hd >> 6, d = hd & 63;
  const float* W = (s == 0) ? Wq : ((s == 1) ? Wk : Wv);
  out[idx] = (bf16)W[((size_t)h * CC + c) * HS + d];
}

// W [K,N] row-major (f32) -> Wt [N,K] bf16
__global__ __launch_bounds__(256)
void transpose_w_kernel(const float* __restrict__ W, bf16* __restrict__ Wt,
                        int K, int N) {
  size_t idx = (size_t)blockIdx.x * 256 + threadIdx.x;
  if (idx >= (size_t)K * N) return;
  int n = (int)(idx / K);
  int k = (int)(idx % K);
  Wt[idx] = (bf16)W[(size_t)k * N + n];
}

// Vt[((b*H+h)*64+d)*T + t] = qkv[(b*T+t)*3072 + 2048 + h*64 + d]
__global__ __launch_bounds__(256)
void transpose_v_kernel(const bf16* __restrict__ qkv, bf16* __restrict__ Vt) {
  size_t idx = (size_t)blockIdx.x * 256 + threadIdx.x;
  if (idx >= (size_t)BB * HH * HS * TT) return;
  int t = (int)(idx % TT);
  size_t rest = idx / TT;
  int d = (int)(rest % HS);
  size_t bh = rest / HS;
  int h = (int)(bh % HH);
  int b = (int)(bh / HH);
  Vt[idx] = qkv[((size_t)(b * TT + t)) * C3 + 2 * CC + h * HS + d];
}

// ---------------- GEMM: D = A * Bt^T (+bias)(+relu)(+resid) ----------------
// A:  [M,K] bf16 row-major     Bt: [N,K] bf16 row-major
// MODE 0: bf16 out             MODE 1: bf16 out, +bias, relu
// MODE 2: f32 out, +bias, +resid
// Block: 256 thr = 8 waves, block tile 64(M) x 256(N), k-step 32.
// A/B tiles staged in double-buffered LDS via global_load_async_to_lds_b128;
// each wave issues exactly 5 async ops per group, so `s_wait_asynccnt 5`
// after issuing the next group completes the current one (in-order).
template <int MODE>
__global__ __launch_bounds__(256)
void gemm_bt_kernel(const bf16* __restrict__ A, const bf16* __restrict__ Bt,
                    const float* __restrict__ bias, const float* __restrict__ resid,
                    void* __restrict__ outv, int M, int N, int K) {
  __shared__ __align__(16) bf16 sA[2][64 * 32];    // 2 x 4 KB
  __shared__ __align__(16) bf16 sB[2][256 * 32];   // 2 x 16 KB
  int tid  = threadIdx.x;
  int lane = tid & 31;
  int wave = tid >> 5;
  int wm = wave & 1, wn = wave >> 1;
  int rowBase = blockIdx.y * 64;
  int colBase = blockIdx.x * 256;
  int lhalf = lane >> 4;   // 0|1
  int lmod  = lane & 15;

  // async-copy assignment: A tile 64x32 = 256 chunks of 16B (1/thread),
  // B tile 256x32 = 1024 chunks (4/thread)
  int ar = tid >> 2;              // A row 0..63
  int ac = (tid & 3) * 8;         // A col 0,8,16,24
  const bf16* gA = A + (size_t)(rowBase + ar) * K + ac;
  const bf16* gB[4];
  uint32_t laOff[2], lbOff[2][4];
#pragma unroll
  for (int j = 0; j < 4; ++j) {
    int n = (tid >> 2) + j * 64;  // B row (n) 0..255
    gB[j] = Bt + (size_t)(colBase + n) * K + ac;
  }
#pragma unroll
  for (int u = 0; u < 2; ++u) {
    laOff[u] = lds_off(&sA[u][ar * 32 + ac]);
#pragma unroll
    for (int j = 0; j < 4; ++j)
      lbOff[u][j] = lds_off(&sB[u][((tid >> 2) + j * 64) * 32 + ac]);
  }

  auto issue = [&](int buf, int k0) {
    async_b128(gA + k0, laOff[buf]);
#pragma unroll
    for (int j = 0; j < 4; ++j) async_b128(gB[j] + k0, lbOff[buf][j]);
  };

  const int nsteps = K / 32;
  v8f acc[2][4] = {};
  issue(0, 0);
  for (int i = 0; i < nsteps; ++i) {
    int cur = i & 1;
    __syncthreads();                       // buf cur^1 fully consumed -> reusable
    if (i + 1 < nsteps) {
      issue(cur ^ 1, (i + 1) * 32);
      asm volatile("s_wait_asynccnt 5" ::: "memory");   // group i done
    } else {
      asm volatile("s_wait_asynccnt 0" ::: "memory");
    }
    __syncthreads();                       // all waves' tile writes visible

    V16 a[2], b[4];
#pragma unroll
    for (int mt = 0; mt < 2; ++mt) {
      const bf16* p = &sA[cur][(wm * 32 + mt * 16 + lmod) * 32 + lhalf * 8];
      a[mt].h[0] = *(const v8bf*)p;
      a[mt].h[1] = *(const v8bf*)(p + 16);
    }
#pragma unroll
    for (int nt = 0; nt < 4; ++nt) {
      const bf16* p = &sB[cur][(wn * 64 + nt * 16 + lmod) * 32 + lhalf * 8];
      b[nt].h[0] = *(const v8bf*)p;
      b[nt].h[1] = *(const v8bf*)(p + 16);
    }
#pragma unroll
    for (int mt = 0; mt < 2; ++mt)
#pragma unroll
      for (int nt = 0; nt < 4; ++nt)
        acc[mt][nt] = wmma_bf16(a[mt].v, b[nt].v, acc[mt][nt]);
  }

  // epilogue: lane holds col (lmod), rows lhalf*8 + r of each 16x16 tile
#pragma unroll
  for (int mt = 0; mt < 2; ++mt) {
#pragma unroll
    for (int nt = 0; nt < 4; ++nt) {
      int col = colBase + wn * 64 + nt * 16 + lmod;
      float bv = (MODE == 0) ? 0.f : bias[col];
#pragma unroll
      for (int r = 0; r < 8; ++r) {
        int row = rowBase + wm * 32 + mt * 16 + lhalf * 8 + r;
        float v = acc[mt][nt][r] + bv;
        if (MODE == 1) v = v > 0.f ? v : 0.f;
        if (MODE == 2) {
          float* out = (float*)outv;
          out[(size_t)row * N + col] = v + resid[(size_t)row * N + col];
        } else {
          bf16* out = (bf16*)outv;
          out[(size_t)row * N + col] = (bf16)v;
        }
      }
    }
  }
}

// ---------------- flash attention (causal) ----------------
// 1 wave per (b,h,16-query tile). qkv: [M,3072] bf16; Vt: [B,H,64,T] bf16;
// out: attn [M, C] bf16 with head-concat layout col = h*64+d.
__global__ __launch_bounds__(32)
void attn_kernel(const bf16* __restrict__ qkv, const bf16* __restrict__ Vt,
                 bf16* __restrict__ attnOut) {
  int qi = blockIdx.x;
  int h  = blockIdx.y;
  int b  = blockIdx.z;
  int lane  = threadIdx.x;
  int lhalf = lane >> 4;
  int lmod  = lane & 15;
  int t0 = qi * 16;
  __shared__ __align__(16) bf16 Pz[16 * 32];

  // resident Q fragments (hs=64 => 2 k-steps)
  V16 qf[2];
  {
    const bf16* qbase = qkv + ((size_t)(b * TT + t0 + lmod)) * C3 + h * HS;
#pragma unroll
    for (int ks = 0; ks < 2; ++ks) {
      const bf16* p = qbase + ks * 32 + lhalf * 8;
      qf[ks].h[0] = *(const v8bf*)p;
      qf[ks].h[1] = *(const v8bf*)(p + 16);
    }
  }

  v8f O[4] = {};
  float mrow[8], lrow[8];
#pragma unroll
  for (int r = 0; r < 8; ++r) { mrow[r] = -3.0e38f; lrow[r] = 0.f; }
  const float scale = 0.03125f;  // C^-0.5 = 1024^-0.5 (reference scales by full C)
  const bf16* vbase = Vt + ((size_t)(b * HH + h)) * HS * TT;

  for (int kb = 0; kb <= t0 + 15; kb += 32) {
    // two 16x16 score tiles: keys [kb,kb+16) and [kb+16,kb+32)
    v8f sa = {}, sb = {};
#pragma unroll
    for (int ks = 0; ks < 2; ++ks) {
      V16 kfa, kfb;
      const bf16* p = qkv + ((size_t)(b * TT + kb + lmod)) * C3 + CC + h * HS
                      + ks * 32 + lhalf * 8;
      kfa.h[0] = *(const v8bf*)p;       kfa.h[1] = *(const v8bf*)(p + 16);
      const bf16* p2 = p + (size_t)16 * C3;
      kfb.h[0] = *(const v8bf*)p2;      kfb.h[1] = *(const v8bf*)(p2 + 16);
      sa = wmma_bf16(qf[ks].v, kfa.v, sa);
      sb = wmma_bf16(qf[ks].v, kfb.v, sb);
    }
    // scale + causal mask + local row max (lane's column is key index)
    float mloc[8];
#pragma unroll
    for (int r = 0; r < 8; ++r) {
      int row = t0 + lhalf * 8 + r;
      float va = sa[r] * scale; if (kb + lmod > row)      va = -3.0e38f;
      float vb = sb[r] * scale; if (kb + 16 + lmod > row) vb = -3.0e38f;
      sa[r] = va; sb[r] = vb;
      mloc[r] = fmaxf(va, vb);
    }
#pragma unroll
    for (int off = 1; off < 16; off <<= 1)
#pragma unroll
      for (int r = 0; r < 8; ++r) mloc[r] = fmaxf(mloc[r], __shfl_xor(mloc[r], off, 32));
    float alpha[8];
#pragma unroll
    for (int r = 0; r < 8; ++r) {
      float mnew = fmaxf(mrow[r], mloc[r]);
      alpha[r] = __expf(mrow[r] - mnew);
      mrow[r] = mnew;
    }
    float ls[8];
#pragma unroll
    for (int r = 0; r < 8; ++r) {
      float pa = __expf(sa[r] - mrow[r]);
      float pb = __expf(sb[r] - mrow[r]);
      sa[r] = pa; sb[r] = pb;
      ls[r] = pa + pb;
    }
#pragma unroll
    for (int off = 1; off < 16; off <<= 1)
#pragma unroll
      for (int r = 0; r < 8; ++r) ls[r] += __shfl_xor(ls[r], off, 32);
#pragma unroll
    for (int r = 0; r < 8; ++r) lrow[r] = lrow[r] * alpha[r] + ls[r];
#pragma unroll
    for (int dt = 0; dt < 4; ++dt)
#pragma unroll
      for (int r = 0; r < 8; ++r) O[dt][r] *= alpha[r];

    // stage P (C-layout -> row-major 16x32 bf16 in LDS -> A fragment)
    __syncthreads();  // single-wave WG: barrier is a NOP, keeps LDS order
#pragma unroll
    for (int r = 0; r < 8; ++r) {
      Pz[(lhalf * 8 + r) * 32 + lmod]      = (bf16)sa[r];
      Pz[(lhalf * 8 + r) * 32 + 16 + lmod] = (bf16)sb[r];
    }
    __syncthreads();
    V16 pf;
    {
      const bf16* p = &Pz[lmod * 32 + lhalf * 8];
      pf.h[0] = *(const v8bf*)p;
      pf.h[1] = *(const v8bf*)(p + 16);
    }
#pragma unroll
    for (int dt = 0; dt < 4; ++dt) {
      V16 vf;
      const bf16* p = vbase + ((size_t)(dt * 16 + lmod)) * TT + kb + lhalf * 8;
      vf.h[0] = *(const v8bf*)p;
      vf.h[1] = *(const v8bf*)(p + 16);
      O[dt] = wmma_bf16(pf.v, vf.v, O[dt]);
    }
  }
#pragma unroll
  for (int dt = 0; dt < 4; ++dt)
#pragma unroll
    for (int r = 0; r < 8; ++r) {
      int row = t0 + lhalf * 8 + r;
      attnOut[((size_t)(b * TT + row)) * CC + h * HS + dt * 16 + lmod] =
          (bf16)(O[dt][r] / lrow[r]);
    }
}

// ---------------- orchestration ----------------
extern "C" void kernel_launch(void* const* d_in, const int* in_sizes, int n_in,
                              void* d_out, int out_size, void* d_ws, size_t ws_size,
                              hipStream_t stream) {
  (void)in_sizes; (void)n_in; (void)out_size; (void)ws_size;
  const float* x     = (const float*)d_in[0];
  const float* Wq    = (const float*)d_in[1];
  const float* Wk    = (const float*)d_in[2];
  const float* Wv    = (const float*)d_in[3];
  const float* Wproj = (const float*)d_in[4];
  const float* bproj = (const float*)d_in[5];
  const float* ln1_g = (const float*)d_in[6];
  const float* ln1_b = (const float*)d_in[7];
  const float* ln2_g = (const float*)d_in[8];
  const float* ln2_b = (const float*)d_in[9];
  const float* W1    = (const float*)d_in[10];
  const float* b1    = (const float*)d_in[11];
  const float* W2    = (const float*)d_in[12];
  const float* b2    = (const float*)d_in[13];
  float* out = (float*)d_out;

  char* ws = (char*)d_ws;
  size_t off = 0;
  auto alloc = [&](size_t bytes) {
    char* p = ws + off;
    off = (off + bytes + 255) & ~(size_t)255;
    return p;
  };
  bf16*  h1   = (bf16*)alloc((size_t)MM * CC * 2);
  bf16*  qkvW = (bf16*)alloc((size_t)C3 * CC * 2);
  bf16*  qkv  = (bf16*)alloc((size_t)MM * C3 * 2);
  bf16*  Vt   = (bf16*)alloc((size_t)BB * HH * HS * TT * 2);
  bf16*  attn = (bf16*)alloc((size_t)MM * CC * 2);
  bf16*  Wpt  = (bf16*)alloc((size_t)CC * CC * 2);
  float* x2   = (float*)alloc((size_t)MM * CC * 4);
  bf16*  h2   = (bf16*)alloc((size_t)MM * CC * 2);
  bf16*  W1t  = (bf16*)alloc((size_t)CC * FF * 2);
  bf16*  f1   = (bf16*)alloc((size_t)MM * FF * 2);
  bf16*  W2t  = (bf16*)alloc((size_t)FF * CC * 2);

  auto cdiv = [](size_t a, size_t b) { return (unsigned)((a + b - 1) / b); };

  // 1) h1 = bf16(ln1(x))
  layernorm_bf16_kernel<<<MM, 256, 0, stream>>>(x, ln1_g, ln1_b, h1);
  // 2) weight prep (bf16, transposed [N,K])
  pack_qkv_w_kernel<<<cdiv((size_t)C3 * CC, 256), 256, 0, stream>>>(Wq, Wk, Wv, qkvW);
  transpose_w_kernel<<<cdiv((size_t)CC * CC, 256), 256, 0, stream>>>(Wproj, Wpt, CC, CC);
  transpose_w_kernel<<<cdiv((size_t)CC * FF, 256), 256, 0, stream>>>(W1, W1t, CC, FF);
  transpose_w_kernel<<<cdiv((size_t)FF * CC, 256), 256, 0, stream>>>(W2, W2t, FF, CC);
  // 3) qkv = h1 @ qkvW^T            [8192 x 3072]
  gemm_bt_kernel<0><<<dim3(C3 / 256, MM / 64), 256, 0, stream>>>(
      h1, qkvW, nullptr, nullptr, qkv, MM, C3, CC);
  // 4) Vt = transpose(V)
  transpose_v_kernel<<<cdiv((size_t)BB * HH * HS * TT, 256), 256, 0, stream>>>(qkv, Vt);
  // 5) attention
  attn_kernel<<<dim3(TT / 16, HH, BB), 32, 0, stream>>>(qkv, Vt, attn);
  // 6) x2 = x + attn @ Wproj + bproj
  gemm_bt_kernel<2><<<dim3(CC / 256, MM / 64), 256, 0, stream>>>(
      attn, Wpt, bproj, x, x2, MM, CC, CC);
  // 7) h2 = bf16(ln2(x2))
  layernorm_bf16_kernel<<<MM, 256, 0, stream>>>(x2, ln2_g, ln2_b, h2);
  // 8) f1 = relu(h2 @ W1 + b1)      [8192 x 4096]
  gemm_bt_kernel<1><<<dim3(FF / 256, MM / 64), 256, 0, stream>>>(
      h2, W1t, b1, nullptr, f1, MM, FF, CC);
  // 9) out = x2 + f1 @ W2 + b2
  gemm_bt_kernel<2><<<dim3(CC / 256, MM / 64), 256, 0, stream>>>(
      f1, W2t, b2, x2, out, MM, CC, FF);
}
